// ParallelGRUDWrapper_29703993819348
// MI455X (gfx1250) — compile-verified, matrix operand order checked by reading, not run on hardware
//
#include <hip/hip_runtime.h>
#include <hip/hip_bf16.h>
#include <math.h>

// ---------------------------------------------------------------------------
// GRU-D forward for MI455X (gfx1250): wave32 + v_wmma_f32_16x16x32_f16.
// Grid = (nf=64, B/16=8); block = 32 threads (one wave). Each wave owns a
// 16-row batch stripe of one filter and runs the full T=100 recurrence with
// all state (comb [16x264] f16, h [16x132] f32) in LDS. Weights are converted
// once to f16 into d_ws; per-step B-matrix fragments stream from L2 (13 MB
// working set << 192 MB L2) via explicit addrspace(1) pointers so they lower
// to global_load_b128 (SGPR base + VGPR offset, LOADcnt only), and the bases
// are laundered through an asm barrier inside the T loop so LICM cannot hoist
// the loop-invariant fragment loads and spill them to scratch.
// ---------------------------------------------------------------------------

typedef __attribute__((ext_vector_type(16))) _Float16 v16h;
typedef __attribute__((ext_vector_type(8)))  _Float16 v8h;
typedef __attribute__((ext_vector_type(8)))  float    v8f;

// Global-address-space views for the f16 weight array.
typedef __attribute__((address_space(1))) const _Float16 GHalf;
typedef __attribute__((address_space(1))) const v8h      GV8h;

#define NF 64
#define BB 128
#define TT 100
#define II 64
#define HH 128
#define CC 256            // 2*I + H

// param layout offsets (floats) inside each filter's row of `params`
#define OFF_ZW   0
#define OFF_ZB   32768
#define OFF_RW   32896
#define OFF_RB   65664
#define OFF_HW   65792
#define OFF_HB   98560
#define OFF_GXW  98688
#define OFF_GXB  102784
#define OFF_GHW  102848
#define OFF_GHB  111040
#define OFF_FCW  111168
#define OFF_FCB  111296
#define P_TOT    111297

#define PERF_H   (3 * HH * CC + HH * II)   // 106496 halves of f16 weights per filter

#define COMB_S   264   // comb row stride (halves): 132 dwords, gcd(132,64)=4
#define DBUF_S   72    // dbuf row stride (halves): 36 dwords, gcd(36,64)=4
#define HBUF_S   132   // h row stride (floats): gcd(132,64)=4

// ---------------------------------------------------------------------------
// Pre-pass: convert zw/rw/hw/ghw to f16 into d_ws (row-major, per-filter).
// ---------------------------------------------------------------------------
__global__ void grud_convert_weights(const float* __restrict__ params,
                                     _Float16* __restrict__ wh) {
  const long long total = (long long)NF * PERF_H;
  for (long long idx = (long long)blockIdx.x * blockDim.x + threadIdx.x;
       idx < total; idx += (long long)gridDim.x * blockDim.x) {
    int n = (int)(idx / PERF_H);
    int j = (int)(idx % PERF_H);
    const float* p = params + (long long)n * P_TOT;
    float v;
    if (j < HH * CC)           v = p[OFF_ZW + j];
    else if (j < 2 * HH * CC)  v = p[OFF_RW + (j - HH * CC)];
    else if (j < 3 * HH * CC)  v = p[OFF_HW + (j - 2 * HH * CC)];
    else                       v = p[OFF_GHW + (j - 3 * HH * CC)];
    wh[idx] = (_Float16)v;
  }
}

// ---------------------------------------------------------------------------
// WMMA fragment loaders. 16-bit A-layout (ISA 7.12.2): lane = row M (bits 3:0),
// lane bit 4 = K-half selector g; the lane's 16 halves are two contiguous
// 8-half runs at K = 8g and K = 16+8g. B mirrors A with lane = column N.
// ---------------------------------------------------------------------------
__device__ __forceinline__ v16h load_fragA(const _Float16* base, int stride,
                                           int kbase, int row, int g) {
  const _Float16* p = base + (size_t)row * stride + kbase + 8 * g;
  v8h lo = *(const v8h*)(p);        // K = kbase + 8g .. +7
  v8h hi = *(const v8h*)(p + 16);   // K = kbase + 16 + 8g .. +7
  return __builtin_shufflevector(lo, hi, 0, 1, 2, 3, 4, 5, 6, 7,
                                         8, 9, 10, 11, 12, 13, 14, 15);
}

__device__ __forceinline__ v16h load_fragW(const GHalf* base, int stride,
                                           int kbase, int row, int g) {
  const GHalf* p = base + (size_t)row * stride + kbase + 8 * g;
  v8h lo = *(const GV8h*)(p);       // global_load_b128, saddr + voffset
  v8h hi = *(const GV8h*)(p + 16);
  return __builtin_shufflevector(lo, hi, 0, 1, 2, 3, 4, 5, 6, 7,
                                         8, 9, 10, 11, 12, 13, 14, 15);
}

__device__ __forceinline__ v8f vzero8() {
  v8f z;
#pragma unroll
  for (int i = 0; i < 8; ++i) z[i] = 0.0f;
  return z;
}

// acc[t] covers output columns [16t, 16t+16) for this wave's 16 rows.
// A in LDS ([16 x As] halves), W in global ([128 x Ws] halves, row = out col).
template <int KSTEPS>
__device__ __forceinline__ void gemm8(v8f acc[8],
                                      const _Float16* __restrict__ A, int As,
                                      const GHalf* __restrict__ W, int Ws,
                                      int nidx, int g) {
#pragma unroll
  for (int k = 0; k < KSTEPS; ++k) {
    v16h a = load_fragA(A, As, 32 * k, nidx, g);
#pragma unroll
    for (int t = 0; t < 8; ++t) {
      v16h b = load_fragW(W, Ws, 32 * k, 16 * t + nidx, g);
      acc[t] = __builtin_amdgcn_wmma_f32_16x16x32_f16(
          false, a, false, b, (short)0, acc[t], false, false);
    }
  }
}

__device__ __forceinline__ float sigmoidf_(float x) {
  return 1.0f / (1.0f + __expf(-x));
}

// ---------------------------------------------------------------------------
// Main kernel: one wave per (filter, 16-row batch stripe).
// ---------------------------------------------------------------------------
__global__ __launch_bounds__(32) void grud_forward_kernel(
    const float* __restrict__ params, const float* __restrict__ X,
    const float* __restrict__ Xl, const float* __restrict__ xmean,
    const float* __restrict__ Mask, const float* __restrict__ Delta,
    const _Float16* __restrict__ wh, float* __restrict__ out) {
  __shared__ __align__(16) _Float16 comb[16 * COMB_S];  // [x | h | m] f16
  __shared__ __align__(16) _Float16 dbuf[16 * DBUF_S];  // delta f16 (A for dh GEMM)
  __shared__ __align__(16) float    hbuf[16 * HBUF_S];  // h state f32

  const int n    = blockIdx.x;        // filter
  const int b0   = blockIdx.y * 16;   // batch stripe base
  const int lane = threadIdx.x;       // 0..31
  const int nidx = lane & 15;
  const int g    = lane >> 4;

  const float* p = params + (size_t)n * P_TOT;
  const GHalf* zwh0  = (const GHalf*)(wh + (size_t)n * PERF_H);
  const GHalf* rwh0  = zwh0 + HH * CC;
  const GHalf* hwh0  = rwh0 + HH * CC;
  const GHalf* ghwh0 = hwh0 + HH * CC;

  // Per-lane bias registers, indexed by output tile t (col = 16t + nidx).
  float zb_[8], rb_[8], hb_[8], ghb_[8];
#pragma unroll
  for (int t = 0; t < 8; ++t) {
    zb_[t]  = p[OFF_ZB  + 16 * t + nidx];
    rb_[t]  = p[OFF_RB  + 16 * t + nidx];
    hb_[t]  = p[OFF_HB  + 16 * t + nidx];
    ghb_[t] = p[OFF_GHB + 16 * t + nidx];
  }
  // Diagonal FilterLinear params for the two i-columns this lane handles.
  const float gxd0 = p[OFF_GXW + lane * II + lane];
  const float gxd1 = p[OFF_GXW + (lane + 32) * II + (lane + 32)];
  const float gxb0 = p[OFF_GXB + lane];
  const float gxb1 = p[OFF_GXB + lane + 32];

  // h0 = 0
  for (int i = lane; i < 16 * HBUF_S; i += 32) hbuf[i] = 0.0f;
  // Single wave: LDS ops are in-order with later ds reads; no barrier needed.

  for (int t = 0; t < TT; ++t) {
    // Launder the weight bases so their fragment loads stay inside the loop
    // (L2 hits) instead of being LICM-hoisted and spilled to scratch.
    const GHalf* zwh  = zwh0;
    const GHalf* rwh  = rwh0;
    const GHalf* hwh  = hwh0;
    const GHalf* ghwh = ghwh0;
    asm volatile("" : "+s"(zwh), "+s"(rwh), "+s"(hwh), "+s"(ghwh));

    // ---- elementwise input stage: build x, m (comb) and delta (dbuf) -----
    const float* dG  = Delta + (((size_t)n * BB + b0) * TT + t) * II;
    const float* mG  = Mask  + (((size_t)n * BB + b0) * TT + t) * II;
    const float* xG  = X  + ((size_t)b0 * TT + t) * II;
    const float* xlG = Xl + ((size_t)b0 * TT + t) * II;
    const float* xmG = xmean + (size_t)t * II;
    const float xm0 = xmG[lane], xm1 = xmG[lane + 32];

#pragma unroll 4
    for (int r = 0; r < 16; ++r) {
      const size_t ro = (size_t)r * TT * II;  // row stride within [*,T,I]
      float d0  = dG[ro + lane],  d1  = dG[ro + lane + 32];
      float m0  = mG[ro + lane],  m1  = mG[ro + lane + 32];
      float xt0 = xG[ro + lane],  xt1 = xG[ro + lane + 32];
      float xl0 = xlG[ro + lane], xl1 = xlG[ro + lane + 32];
      float dx0 = __expf(-fmaxf(0.0f, d0 * gxd0 + gxb0));
      float dx1 = __expf(-fmaxf(0.0f, d1 * gxd1 + gxb1));
      float xv0 = m0 * xt0 + (1.0f - m0) * (dx0 * xl0 + (1.0f - dx0) * xm0);
      float xv1 = m1 * xt1 + (1.0f - m1) * (dx1 * xl1 + (1.0f - dx1) * xm1);
      comb[r * COMB_S + lane]            = (_Float16)xv0;   // x cols [0,64)
      comb[r * COMB_S + lane + 32]       = (_Float16)xv1;
      comb[r * COMB_S + 192 + lane]      = (_Float16)m0;    // m cols [192,256)
      comb[r * COMB_S + 192 + lane + 32] = (_Float16)m1;
      dbuf[r * DBUF_S + lane]            = (_Float16)d0;
      dbuf[r * DBUF_S + lane + 32]       = (_Float16)d1;
    }

    // ---- delta_h = exp(-relu(d @ ghw^T + ghb)); h *= delta_h -------------
    v8f dha[8];
#pragma unroll
    for (int i = 0; i < 8; ++i) dha[i] = vzero8();
    gemm8<2>(dha, dbuf, DBUF_S, ghwh, II, nidx, g);
#pragma unroll
    for (int tt = 0; tt < 8; ++tt) {
#pragma unroll
      for (int v = 0; v < 8; ++v) {
        const int mrow = v + 8 * g;
        const int col  = 16 * tt + nidx;
        float dh = __expf(-fmaxf(0.0f, dha[tt][v] + ghb_[tt]));
        float hv = hbuf[mrow * HBUF_S + col] * dh;
        hbuf[mrow * HBUF_S + col]       = hv;
        comb[mrow * COMB_S + 64 + col]  = (_Float16)hv;     // h cols [64,192)
      }
    }

    // ---- z = sigmoid(comb @ zw^T + zb) (kept in registers) ---------------
    v8f zacc[8];
#pragma unroll
    for (int i = 0; i < 8; ++i) zacc[i] = vzero8();
    gemm8<8>(zacc, comb, COMB_S, zwh, CC, nidx, g);
#pragma unroll
    for (int tt = 0; tt < 8; ++tt)
#pragma unroll
      for (int v = 0; v < 8; ++v)
        zacc[tt][v] = sigmoidf_(zacc[tt][v] + zb_[tt]);

    // ---- r = sigmoid(comb @ rw^T + rb); comb.h <- r*h --------------------
    {
      v8f racc[8];
#pragma unroll
      for (int i = 0; i < 8; ++i) racc[i] = vzero8();
      gemm8<8>(racc, comb, COMB_S, rwh, CC, nidx, g);
#pragma unroll
      for (int tt = 0; tt < 8; ++tt) {
#pragma unroll
        for (int v = 0; v < 8; ++v) {
          const int mrow = v + 8 * g;
          const int col  = 16 * tt + nidx;
          float rr = sigmoidf_(racc[tt][v] + rb_[tt]);
          float hv = hbuf[mrow * HBUF_S + col];
          comb[mrow * COMB_S + 64 + col] = (_Float16)(rr * hv);
        }
      }
    }

    // ---- h_tilde = tanh(comb2 @ hw^T + hb); h = (1-z)h + z*h_tilde -------
    {
      v8f hacc[8];
#pragma unroll
      for (int i = 0; i < 8; ++i) hacc[i] = vzero8();
      gemm8<8>(hacc, comb, COMB_S, hwh, CC, nidx, g);
#pragma unroll
      for (int tt = 0; tt < 8; ++tt) {
#pragma unroll
        for (int v = 0; v < 8; ++v) {
          const int mrow = v + 8 * g;
          const int col  = 16 * tt + nidx;
          float ht = tanhf(hacc[tt][v] + hb_[tt]);
          float zz = zacc[tt][v];
          float hv = hbuf[mrow * HBUF_S + col];
          hbuf[mrow * HBUF_S + col] = (1.0f - zz) * hv + zz * ht;
        }
      }
    }
  }

  // ---- final FC: out = sigmoid(h . fcw + fcb) ----------------------------
  const float* fcw = p + OFF_FCW;
  const float  fcb = p[OFF_FCB];
  float s = 0.0f;
#pragma unroll 8
  for (int c = 0; c < 64; ++c)
    s += hbuf[nidx * HBUF_S + 64 * g + c] * fcw[64 * g + c];
  s += __shfl_xor(s, 16, 32);  // combine the two half-row partials
  if (g == 0)
    out[(size_t)n * BB + b0 + nidx] = sigmoidf_(s + fcb);
}

// ---------------------------------------------------------------------------
extern "C" void kernel_launch(void* const* d_in, const int* in_sizes, int n_in,
                              void* d_out, int out_size, void* d_ws, size_t ws_size,
                              hipStream_t stream) {
  (void)in_sizes; (void)n_in; (void)out_size; (void)ws_size;
  const float* params = (const float*)d_in[0];
  const float* X      = (const float*)d_in[1];
  const float* Xl     = (const float*)d_in[2];
  const float* xmean  = (const float*)d_in[3];
  const float* Mask   = (const float*)d_in[4];
  const float* Delta  = (const float*)d_in[5];
  _Float16* wh = (_Float16*)d_ws;   // 13 MB of f16 weights, lives in L2

  grud_convert_weights<<<512, 256, 0, stream>>>(params, wh);

  dim3 grid(NF, BB / 16);           // 64 filters x 8 batch stripes = 512 waves
  grud_forward_kernel<<<grid, 32, 0, stream>>>(params, X, Xl, xmean, Mask,
                                               Delta, wh, (float*)d_out);
}
